// GeneralAttention_72748156059916
// MI455X (gfx1250) — matrix operand order repaired and checked by reading, not compiled, not run on hardware
//
#include <hip/hip_runtime.h>
#include <math.h>

// ---------------------------------------------------------------------------
// GeneralAttention on MI455X (gfx1250):
//   L = x@W1 ; R = x@W3 ; M = L@W2 ; S = M@R^T + bias ; attn = softmax(S) ;
//   out = attn @ V_f.   N = DIM = 4096, fp32 in/out.
//
// Strategy:
//   * One-time pass converts x to bf16 and W1/W2/W3/V_f to *transposed* bf16.
//   * Every GEMM is then  C = A[M,K] * B^T  with both operands bf16, K-major.
//   * GEMM tiles staged via GLOBAL_LOAD_ASYNC_TO_LDS_B128 (ASYNCcnt),
//     double-buffered; compute = V_WMMA_F32_16X16X32_BF16, f32 accumulators.
//   * 128x256 block tile, 8 waves (2x4), 64x64 per wave = 4x4 WMMA tiles
//     -> 16 WMMAs per 16 ds_load_b128 per K-step, 4x fragment reuse.
// ---------------------------------------------------------------------------

typedef __bf16 bf16;
typedef __attribute__((ext_vector_type(16))) __bf16 v16bf;
typedef __attribute__((ext_vector_type(8)))  float  v8f;

constexpr int BM = 128;        // block tile M
constexpr int BN = 256;        // block tile N
constexpr int BK = 32;         // K step = one WMMA
constexpr int KS = BK + 8;     // padded LDS k-stride: 80 B rows (16B aligned)

constexpr unsigned A_TILE_BYTES = (unsigned)(BM * KS * 2);   // 10240
constexpr unsigned B_TILE_BYTES = (unsigned)(BN * KS * 2);   // 20480

// ---------------------------------------------------------------------------
// Async DMA of a ROWS x 32 bf16 tile (K-contiguous rows) into LDS.
// gsrc points at (row0, k0); row stride = ldk elements. Each row = 4 x 16B
// chunks; 256 threads issue ROWS*4/256 async B128 each.
// ---------------------------------------------------------------------------
template<int ROWS>
__device__ __forceinline__ void stage_tile_async(const bf16* gsrc, int ldk,
                                                 unsigned lds_base)
{
    const int tid = threadIdx.x;
    #pragma unroll
    for (int i = 0; i < (ROWS * 4) / 256; ++i) {
        const int chunk = tid + i * 256;
        const int r = chunk >> 2;             // row
        const int c = chunk & 3;              // 16B chunk within 64B row
        const unsigned voff = (unsigned)(r * ldk * 2 + c * 16);
        const unsigned loff = lds_base + (unsigned)(r * (KS * 2) + c * 16);
        asm volatile("global_load_async_to_lds_b128 %0, %1, %2"
                     :: "v"(loff), "v"(voff), "s"(gsrc) : "memory");
    }
}

// ---------------------------------------------------------------------------
// C[M,N] = A[M,K] * B^T   (A: bf16 [M,K], Bt: bf16 [N,K], both row-major)
// 256 threads = 8 waves in a 2(M) x 4(N) grid; 64x64 per wave
// => 4x4 WMMA 16x16 f32 accumulator tiles (128 acc VGPRs).
// ---------------------------------------------------------------------------
template<typename TC, bool ADD_BIAS>
__global__ __launch_bounds__(256)
void gemm_bf16_nt_kernel(const bf16* __restrict__ A, const bf16* __restrict__ Bt,
                         const float* __restrict__ bias, TC* __restrict__ C,
                         int M, int N, int K)
{
    __shared__ bf16 As[2][BM][KS];
    __shared__ bf16 Bs[2][BN][KS];

    const int tid   = threadIdx.x;
    const int lane  = tid & 31;
    const int wave  = tid >> 5;
    const int wm0   = (wave >> 2) * 64;   // wave row origin in block tile
    const int wn0   = (wave & 3)  * 64;   // wave col origin in block tile
    const int khalf = lane >> 4;          // 0: lanes 0-15, 1: lanes 16-31
    const int l16   = lane & 15;

    const int bm0 = blockIdx.y * BM;
    const int bn0 = blockIdx.x * BN;

    const unsigned as_base = (unsigned)(uintptr_t)&As[0][0][0];
    const unsigned bs_base = (unsigned)(uintptr_t)&Bs[0][0][0];

    const bf16* Ab = A  + (size_t)bm0 * K;
    const bf16* Bb = Bt + (size_t)bn0 * K;

    v8f acc[4][4];
    #pragma unroll
    for (int i = 0; i < 4; ++i)
        #pragma unroll
        for (int j = 0; j < 4; ++j)
            acc[i][j] = (v8f)0.0f;

    // prologue: stage tile 0 into buffer 0 (2+4 async B128 per wave)
    stage_tile_async<BM>(Ab, K, as_base);
    stage_tile_async<BN>(Bb, K, bs_base);

    const int KT = K / BK;
    for (int kt = 0; kt < KT; ++kt) {
        const int cur = kt & 1;
        if (kt + 1 < KT) {
            const int nxt = cur ^ 1;
            stage_tile_async<BM>(Ab + (size_t)(kt + 1) * BK, K,
                                 as_base + (unsigned)nxt * A_TILE_BYTES);
            stage_tile_async<BN>(Bb + (size_t)(kt + 1) * BK, K,
                                 bs_base + (unsigned)nxt * B_TILE_BYTES);
            // 6 next-tile copies in flight; current tile guaranteed arrived
            asm volatile("s_wait_asynccnt 0x6" ::: "memory");
        } else {
            asm volatile("s_wait_asynccnt 0x0" ::: "memory");
        }
        __syncthreads();

        // ---- B fragments (32x16): lanes 0-15 K=0..15, lanes 16-31 K=16..31
        v16bf bfrag[4];
        #pragma unroll
        for (int tj = 0; tj < 4; ++tj) {
            const bf16* bp = &Bs[cur][wn0 + tj * 16 + l16][khalf * 16];
            #pragma unroll
            for (int i = 0; i < 16; ++i) bfrag[tj][i] = bp[i];
        }
        // ---- A fragments (16x32): elems 0..15 -> K=16*khalf+i
        #pragma unroll
        for (int ti = 0; ti < 4; ++ti) {
            const bf16* ap = &As[cur][wm0 + ti * 16 + l16][0];
            v16bf af;
            #pragma unroll
            for (int i = 0; i < 16; ++i) {
                af[i] = ap[khalf * 16 + i];
            }
            #pragma unroll
            for (int tj = 0; tj < 4; ++tj)
                acc[ti][tj] = __builtin_amdgcn_wmma_f32_16x16x32_bf16(
                    false, af, false, bfrag[tj],
                    (short)0, acc[ti][tj], false, false);
        }
        __syncthreads();
    }

    // ---- epilogue: C/D layout: VGPR r -> row = r + 8*khalf, col = lane&15
    #pragma unroll
    for (int ti = 0; ti < 4; ++ti) {
        #pragma unroll
        for (int tj = 0; tj < 4; ++tj) {
            const int gcol = bn0 + wn0 + tj * 16 + l16;
            const float badd = ADD_BIAS ? bias[gcol] : 0.0f;
            #pragma unroll
            for (int r = 0; r < 8; ++r) {
                const int grow = bm0 + wm0 + ti * 16 + r + 8 * khalf;
                C[(size_t)grow * N + gcol] = (TC)(acc[ti][tj][r] + badd);
            }
        }
    }
}

// ---------------------------------------------------------------------------
// Elementwise fp32 -> bf16 (x). 1024 elements per block.
// ---------------------------------------------------------------------------
__global__ __launch_bounds__(256)
void convert_bf16_kernel(const float* __restrict__ in, bf16* __restrict__ out)
{
    const size_t base = (size_t)blockIdx.x * 1024 + threadIdx.x;
    #pragma unroll
    for (int j = 0; j < 4; ++j)
        out[base + j * 256] = (bf16)in[base + j * 256];
}

// ---------------------------------------------------------------------------
// fp32 -> bf16 with transpose: out = in^T (n x n), 64x64 LDS-staged tiles.
// ---------------------------------------------------------------------------
__global__ __launch_bounds__(256)
void transpose_bf16_kernel(const float* __restrict__ in, bf16* __restrict__ out,
                           int n)
{
    __shared__ bf16 t[64][64 + 8];
    const int tid = threadIdx.x;
    const int r0 = blockIdx.y * 64, c0 = blockIdx.x * 64;
    #pragma unroll
    for (int i = 0; i < 16; ++i) {
        const int e = tid + i * 256;
        const int r = e >> 6, c = e & 63;
        t[c][r] = (bf16)in[(size_t)(r0 + r) * n + (c0 + c)];
    }
    __syncthreads();
    #pragma unroll
    for (int i = 0; i < 16; ++i) {
        const int e = tid + i * 256;
        const int r = e >> 6, c = e & 63;
        out[(size_t)(c0 + r) * n + (r0 + c)] = t[r][c];
    }
}

// ---------------------------------------------------------------------------
// Row softmax: one 256-thread block per row of S (4096 fp32), writes bf16.
// ---------------------------------------------------------------------------
__global__ __launch_bounds__(256)
void softmax_kernel(const float* __restrict__ S, bf16* __restrict__ attn, int N)
{
    __shared__ float red[256];
    const int row = blockIdx.x;
    const int tid = threadIdx.x;
    const float* srow = S + (size_t)row * N;

    float vals[16];
    float lmax = -INFINITY;
    #pragma unroll
    for (int i = 0; i < 16; ++i) {
        vals[i] = srow[tid + i * 256];
        lmax = fmaxf(lmax, vals[i]);
    }
    red[tid] = lmax;
    __syncthreads();
    for (int s = 128; s > 0; s >>= 1) {
        if (tid < s) red[tid] = fmaxf(red[tid], red[tid + s]);
        __syncthreads();
    }
    const float rmax = red[0];
    __syncthreads();

    float lsum = 0.0f;
    #pragma unroll
    for (int i = 0; i < 16; ++i) {
        vals[i] = __expf(vals[i] - rmax);
        lsum += vals[i];
    }
    red[tid] = lsum;
    __syncthreads();
    for (int s = 128; s > 0; s >>= 1) {
        if (tid < s) red[tid] += red[tid + s];
        __syncthreads();
    }
    const float inv = 1.0f / red[0];

    bf16* arow = attn + (size_t)row * N;
    #pragma unroll
    for (int i = 0; i < 16; ++i)
        arow[tid + i * 256] = (bf16)(vals[i] * inv);
}

// ---------------------------------------------------------------------------
extern "C" void kernel_launch(void* const* d_in, const int* in_sizes, int n_in,
                              void* d_out, int out_size, void* d_ws, size_t ws_size,
                              hipStream_t stream)
{
    const float* x    = (const float*)d_in[0];
    const float* W1   = (const float*)d_in[1];
    const float* W2   = (const float*)d_in[2];
    const float* W3   = (const float*)d_in[3];
    const float* Vf   = (const float*)d_in[4];
    const float* bias = (const float*)d_in[5];
    float* out = (float*)d_out;

    const int    Nn    = 4096;
    const size_t nelem = (size_t)Nn * Nn;

    // Workspace layout (288 MB):
    //   slot0 : xbf   -> reused as Mbf  (x dead after R)
    //   slot1..4 : W1t, W2t, W3t, Vft  (transposed bf16 weights)
    //   slot5 : Lbf   -> reused as attn (L dead after M)
    //   slot6 : Rbf
    //   then  : S fp32 scores (64 MB)
    char* ws = (char*)d_ws;
    bf16*  xbf  = (bf16*)(ws);
    bf16*  W1t  = (bf16*)(ws + nelem * 2);
    bf16*  W2t  = (bf16*)(ws + nelem * 4);
    bf16*  W3t  = (bf16*)(ws + nelem * 6);
    bf16*  Vft  = (bf16*)(ws + nelem * 8);
    bf16*  Lbf  = (bf16*)(ws + nelem * 10);
    bf16*  Rbf  = (bf16*)(ws + nelem * 12);
    float* S    = (float*)(ws + nelem * 14);
    bf16*  Mbf  = xbf;   // x dead after R = x@W3
    bf16*  attn = Lbf;   // L dead after M = L@W2

    dim3 block(256);
    dim3 gridG(Nn / BN, Nn / BM);       // 16 x 32 GEMM blocks
    dim3 gridT(Nn / 64, Nn / 64);       // 64 x 64 transpose blocks

    // ---- one-time convert / transpose pass (bandwidth-bound, ~20us) -------
    convert_bf16_kernel<<<dim3((unsigned)(nelem / 1024)), block, 0, stream>>>(x, xbf);
    transpose_bf16_kernel<<<gridT, block, 0, stream>>>(W1, W1t, Nn);
    transpose_bf16_kernel<<<gridT, block, 0, stream>>>(W2, W2t, Nn);
    transpose_bf16_kernel<<<gridT, block, 0, stream>>>(W3, W3t, Nn);
    transpose_bf16_kernel<<<gridT, block, 0, stream>>>(Vf, Vft, Nn);

    // ---- compute chain (all WMMA + async-LDS GEMMs) ------------------------
    // L = x @ W1
    gemm_bf16_nt_kernel<bf16, false>
        <<<gridG, block, 0, stream>>>(xbf, W1t, nullptr, Lbf, Nn, Nn, Nn);
    // R = x @ W3
    gemm_bf16_nt_kernel<bf16, false>
        <<<gridG, block, 0, stream>>>(xbf, W3t, nullptr, Rbf, Nn, Nn, Nn);
    // M = L @ W2
    gemm_bf16_nt_kernel<bf16, false>
        <<<gridG, block, 0, stream>>>(Lbf, W2t, nullptr, Mbf, Nn, Nn, Nn);
    // S = M @ R^T + bias   (R is naturally [N,K] for the NT kernel)
    gemm_bf16_nt_kernel<float, true>
        <<<gridG, block, 0, stream>>>(Mbf, Rbf, bias, S, Nn, Nn, Nn);
    // attn = softmax(S) rows
    softmax_kernel<<<dim3(Nn), block, 0, stream>>>(S, attn, Nn);
    // out = attn @ V_f
    gemm_bf16_nt_kernel<float, false>
        <<<gridG, block, 0, stream>>>(attn, Vft, nullptr, out, Nn, Nn, Nn);
}
